// EngramModule_76227079569603
// MI455X (gfx1250) — matrix operand order repaired and checked by reading, not compiled, not run on hardware
//
#include <hip/hip_runtime.h>
#include <hip/hip_bf16.h>

typedef __bf16 bf16x4  __attribute__((ext_vector_type(4)));
typedef __bf16 bf16x8  __attribute__((ext_vector_type(8)));
typedef __bf16 bf16x16 __attribute__((ext_vector_type(16)));
typedef float  f32x8   __attribute__((ext_vector_type(8)));

#define SEQ        8192
#define HID        768
#define MEMDIM     768
#define HDIM       96
#define TILE_ROWS  64
#define MT         4      // 16-row M-tiles per workgroup (B reused 4x per load)
#define THREADS    512
#define LDS_STRIDE 776    // 768 + 8 bf16 pad -> conflict-free ds_load_b128

// ---------------- kernel 0: convert W to bf16 (L2-resident operand) --------
__global__ __launch_bounds__(256)
void engram_convert_w(const float* __restrict__ wk, const float* __restrict__ wv,
                      unsigned short* __restrict__ wkb, unsigned short* __restrict__ wvb,
                      int n)
{
    int i = blockIdx.x * 256 + threadIdx.x;
    if (i < n) {
        __bf16 a = (__bf16)wk[i];
        __bf16 b = (__bf16)wv[i];
        wkb[i] = __builtin_bit_cast(unsigned short, a);
        wvb[i] = __builtin_bit_cast(unsigned short, b);
    }
}

// ---------------- kernel 1: hash + gather + dual GEMM + norm + gate --------
__global__ __launch_bounds__(THREADS)
void engram_main(const float* __restrict__ hidden,
                 const int*   __restrict__ ids,
                 const float* __restrict__ emb,
                 const unsigned short* __restrict__ wkb,
                 const unsigned short* __restrict__ wvb,
                 const float* __restrict__ keyw,
                 const float* __restrict__ valw,
                 float* __restrict__ gated)
{
    __shared__ __align__(16) __bf16 smem[TILE_ROWS * LDS_STRIDE];   // ~99 KB A-panel
    __shared__ int   shid[TILE_ROWS][8];
    __shared__ float s_ssq[2][TILE_ROWS];
    __shared__ float s_dot[TILE_ROWS];
    __shared__ float s_gate[TILE_ROWS];
    __shared__ float s_rsv[TILE_ROWS];

    const int tid = threadIdx.x;
    const long long row0 = (long long)blockIdx.x * TILE_ROWS;

    if (tid < TILE_ROWS) { s_ssq[0][tid] = 0.f; s_ssq[1][tid] = 0.f; s_dot[tid] = 0.f; }

    // ---- phase 0: n-gram hashes, one (row, head-group) per thread ----------
    {
        int r = tid >> 3, g = tid & 7;                 // 64 rows x 8 heads = 512
        long long row = row0 + r;
        int s = (int)(row & (SEQ - 1));
        int n = (g < 4) ? 2 : 3;
        int h = (g < 4) ? g : g - 4;
        int idv = 0;
        if (s >= n - 1) {
            long long base = 17 + 10007 + 1543LL * (n + 1) + 8191LL * (h + 1);
            long long mix = 0;
            for (int p = 0; p < n; ++p) {
                long long mult = ((base + 32771LL * (p + 1) + 65537LL * (h + 1) * (p + 1))
                                  % 2147483647LL) * 2 + 1;
                long long tok = (long long)ids[row - (n - 1) + p];
                long long term = tok * mult;
                mix = (p == 0) ? term : (mix ^ term);
            }
            long long off = (base * 2147483647LL + 97LL * (n + h + 1)) % 2147483647LL;
            idv = (int)((mix + off) % 1023LL) + 1;
        }
        shid[r][g] = idv;
    }
    __syncthreads();

    // ---- phase 1: gather embeddings -> bf16 A-panel in LDS -----------------
    // One (row, head) 96-wide slice per thread; float4 loads, bf16x4 stores.
    {
        int r = tid >> 3, head = tid & 7;
        int idv = shid[r][head];
        __bf16* dst = &smem[r * LDS_STRIDE + head * HDIM];
        if (idv != 0) {
            const float* src = emb + (size_t)((head << 10) + idv) * HDIM;
            #pragma unroll
            for (int d = 0; d < HDIM; d += 4) {
                float4 v = *(const float4*)(src + d);
                bf16x4 o = {(__bf16)v.x, (__bf16)v.y, (__bf16)v.z, (__bf16)v.w};
                *(bf16x4*)(dst + d) = o;
            }
        } else {
            bf16x4 z = {(__bf16)0.f, (__bf16)0.f, (__bf16)0.f, (__bf16)0.f};
            #pragma unroll
            for (int d = 0; d < HDIM; d += 4) *(bf16x4*)(dst + d) = z;
        }
    }
    __syncthreads();

    // ---- phase 2: WMMA GEMMs ----------------------------------------------
    // waves 0-7: key, 8-15: value; each wave: 4 M-tiles x 6 N-tiles.
    const int lane = tid & 31;
    const int w    = tid >> 5;
    const int mat  = w >> 3;
    const int ngrp = w & 7;
    const int r16  = lane & 15, hi = lane >> 4;

    const __bf16* __restrict__ bbase =
        (const __bf16*)(mat ? wvb : wkb) + (size_t)(ngrp * 96 + r16) * MEMDIM + 16 * hi;
    const __bf16* __restrict__ abase = &smem[r16 * LDS_STRIDE + 16 * hi];

    f32x8 acc[MT][6];
    #pragma unroll
    for (int mt = 0; mt < MT; ++mt)
        #pragma unroll
        for (int i = 0; i < 6; ++i) {
            f32x8 z = {0.f, 0.f, 0.f, 0.f, 0.f, 0.f, 0.f, 0.f};
            acc[mt][i] = z;
        }

    for (int kt = 0; kt < MEMDIM / 32; ++kt) {
        // 6 B fragments for this k-step, held across all 4 M-tiles (4x reuse)
        bf16x16 bfrag[6];
        #pragma unroll
        for (int i = 0; i < 6; ++i) {
            const __bf16* bp = bbase + i * (16 * MEMDIM) + kt * 32;
            bf16x8 b0 = *(const bf16x8*)bp;
            bf16x8 b1 = *(const bf16x8*)(bp + 8);
            bfrag[i] = __builtin_shufflevector(b0, b1, 0,1,2,3,4,5,6,7,8,9,10,11,12,13,14,15);
        }
        #pragma unroll
        for (int mt = 0; mt < MT; ++mt) {
            const __bf16* ap = abase + mt * (16 * LDS_STRIDE) + kt * 32;
            bf16x8 a0 = *(const bf16x8*)ap;
            bf16x8 a1 = *(const bf16x8*)(ap + 8);
            bf16x16 a = __builtin_shufflevector(a0, a1, 0,1,2,3,4,5,6,7,8,9,10,11,12,13,14,15);
            #pragma unroll
            for (int i = 0; i < 6; ++i)
                acc[mt][i] = __builtin_amdgcn_wmma_f32_16x16x32_bf16(
                    false, a, false, bfrag[i], (short)0, acc[mt][i], false, false);
        }
    }

    // ---- phase 3: per-row reductions (sum y^2; key: hidden . keyw*y) -------
    // acc[mt][i][v] = y[row = mt*16 + v + 8*hi][col = (ngrp*6+i)*16 + r16]
    const float* __restrict__ hbase =
        hidden + (size_t)(row0 + 8 * hi) * HID + ngrp * 96 + r16;
    const float* __restrict__ kwbase = keyw + ngrp * 96 + r16;

    float pssq[MT][8];
    float pdot[MT][8];
    #pragma unroll
    for (int mt = 0; mt < MT; ++mt)
        #pragma unroll
        for (int v = 0; v < 8; ++v) { pssq[mt][v] = 0.f; pdot[mt][v] = 0.f; }

    #pragma unroll
    for (int i = 0; i < 6; ++i) {
        float kw = kwbase[i * 16];
        #pragma unroll
        for (int mt = 0; mt < MT; ++mt)
            #pragma unroll
            for (int v = 0; v < 8; ++v) {
                float y = acc[mt][i][v];
                pssq[mt][v] += y * y;
                if (mat == 0)
                    pdot[mt][v] += hbase[(mt * 16 + v) * HID + i * 16] * kw * y;
            }
    }
    #pragma unroll
    for (int mt = 0; mt < MT; ++mt)
        #pragma unroll
        for (int v = 0; v < 8; ++v) {
            float sv = pssq[mt][v], dv = pdot[mt][v];
            for (int m = 1; m <= 8; m <<= 1) {          // reduce within 16-lane half
                sv += __shfl_xor(sv, m, 32);
                dv += __shfl_xor(dv, m, 32);
            }
            if (r16 == v) {
                atomicAdd(&s_ssq[mat][mt * 16 + v + 8 * hi], sv);
                if (mat == 0) atomicAdd(&s_dot[mt * 16 + v + 8 * hi], dv);
            }
        }
    __syncthreads();

    if (tid < TILE_ROWS) {
        float rsk = rsqrtf(s_ssq[0][tid] * (1.f / (float)HID) + 1e-6f);
        float rsv = rsqrtf(s_ssq[1][tid] * (1.f / (float)HID) + 1e-6f);
        float z = s_dot[tid] * rsk * 0.03608439182435161f;   // 1/sqrt(768)
        s_gate[tid] = 1.f / (1.f + __expf(-z));
        s_rsv[tid] = rsv;
    }
    __syncthreads();

    // ---- phase 4: value waves emit gated = gate * valw * y_v * rsqrt -------
    if (mat == 1) {
        float* __restrict__ gbase =
            gated + (size_t)(row0 + 8 * hi) * HID + ngrp * 96 + r16;
        const float* __restrict__ vwbase = valw + ngrp * 96 + r16;
        #pragma unroll
        for (int i = 0; i < 6; ++i) {
            float vw = vwbase[i * 16];
            #pragma unroll
            for (int mt = 0; mt < MT; ++mt)
                #pragma unroll
                for (int v = 0; v < 8; ++v) {
                    int r = mt * 16 + v + 8 * hi;
                    gbase[(mt * 16 + v) * HID + i * 16] =
                        s_gate[r] * vw * acc[mt][i][v] * s_rsv[r];
                }
        }
    }
}

// ---------------- kernel 2: causal depthwise conv (K=3), ws stays in L2 ----
__global__ __launch_bounds__(256)
void engram_conv(const float* __restrict__ g, const float* __restrict__ cw,
                 float* __restrict__ out)
{
    long long row = blockIdx.x;
    int s = (int)(row & (SEQ - 1));
    for (int c = threadIdx.x; c < HID; c += 256) {
        float x0 = g[row * HID + c];                              // k = 2
        float x1 = (s >= 1) ? g[(row - 1) * HID + c] : 0.f;       // k = 1
        float x2 = (s >= 2) ? g[(row - 2) * HID + c] : 0.f;       // k = 0
        out[row * HID + c] = x2 * cw[c * 3 + 0] + x1 * cw[c * 3 + 1] + x0 * cw[c * 3 + 2];
    }
}

extern "C" void kernel_launch(void* const* d_in, const int* in_sizes, int n_in,
                              void* d_out, int out_size, void* d_ws, size_t ws_size,
                              hipStream_t stream)
{
    const float* hidden = (const float*)d_in[0];
    const int*   ids    = (const int*)  d_in[1];
    const float* emb    = (const float*)d_in[2];
    const float* wk     = (const float*)d_in[3];
    const float* wv     = (const float*)d_in[4];
    const float* keyw   = (const float*)d_in[5];
    const float* valw   = (const float*)d_in[6];
    const float* convw  = (const float*)d_in[7];
    float* out = (float*)d_out;

    const long long BS = in_sizes[1];          // B*S rows
    const int nW = HID * MEMDIM;               // 589824

    char* ws = (char*)d_ws;
    unsigned short* wkb = (unsigned short*)ws;
    unsigned short* wvb = wkb + nW;
    float* gated = (float*)(ws + (size_t)2 * nW * sizeof(unsigned short));

    engram_convert_w<<<dim3((nW + 255) / 256), dim3(256), 0, stream>>>(wk, wv, wkb, wvb, nW);
    engram_main<<<dim3((unsigned)(BS / TILE_ROWS)), dim3(THREADS), 0, stream>>>(
        hidden, ids, emb, wkb, wvb, keyw, valw, gated);
    engram_conv<<<dim3((unsigned)BS), dim3(256), 0, stream>>>(gated, convw, out);
}
